// DynamicMoEHead_49125835931751
// MI455X (gfx1250) — matrix operand order repaired and checked by reading, not compiled, not run on hardware
//
#include <hip/hip_runtime.h>

// ---------------------------------------------------------------------------
// MoE head: out[b,d] = sum_e w[e,b] * ( lrelu(lrelu(x@W1e+b1)@W2e+b2) @ W3e + b3 )
// bf16 WMMA (v_wmma_f32_16x16x32_bf16), f32 accumulate, fused epilogues.
// Tiles staged LDS-side by the Tensor Data Mover (tensor_load_to_lds +
// s_wait_tensorcnt), double buffered; weights pre-transposed to [N,K] bf16.
// ---------------------------------------------------------------------------

typedef __attribute__((ext_vector_type(16))) __bf16 v16bf;
typedef __attribute__((ext_vector_type(8)))  __bf16 v8bf;
typedef __attribute__((ext_vector_type(8)))  float  v8f;
typedef __attribute__((ext_vector_type(4)))  unsigned int u32x4;
typedef __attribute__((ext_vector_type(8)))  int i32x8;
typedef __attribute__((ext_vector_type(4)))  int i32x4;

#define BM 128
#define BN 128
#define BK 32
#define LP 40  // LDS row stride in bf16 elems: 64B data + 16B pad = 80B (20-bank step)

#if defined(__AMDGCN__) && __has_builtin(__builtin_amdgcn_tensor_load_to_lds)
#define HAVE_TDM 1
#else
#define HAVE_TDM 0
#endif

#if HAVE_TDM
typedef __attribute__((address_space(3))) char lds_char;

__device__ __forceinline__ unsigned lds_off(const void* p) {
  return (unsigned)(unsigned long long)(lds_char*)p;  // LDS byte offset
}

// 2-D TDM load: tile (tile0 x tile1) elements (2B each) from row-major tensor
// [tdim1, tdim0] with row stride stride0 (elements), into LDS at ldsOff with
// 16B padding appended to every 64B of data (LDS row stride 80B).
__device__ __forceinline__ void tdm_load_2d(const void* gptr, unsigned ldsOff,
                                            int tdim0, int tdim1,
                                            unsigned long long stride0,
                                            int tile0, int tile1) {
  unsigned long long ga = (unsigned long long)gptr;
  u32x4 g0;
  g0.x = 1u;                                          // count=1, user mode
  g0.y = ldsOff;                                      // lds_addr
  g0.z = (unsigned)(ga & 0xFFFFFFFFull);              // global_addr[31:0]
  g0.w = (unsigned)((ga >> 32) & 0x01FFFFFFull)       // global_addr[56:32]
         | 0x80000000u;                               // type=2 ("image")
  // group1: data_size=1 (2B) | pad_enable | pad_interval=3 (16 DW = 64B)
  //         | pad_amount=3 (4 DW = 16B)
  unsigned w0 = (1u << 16) | (1u << 20) | (3u << 22) | (3u << 25);
  unsigned w1 = ((unsigned)tdim0 & 0xFFFFu) << 16;                      // dim0[15:0]
  unsigned w2 = ((unsigned)tdim0 >> 16) | (((unsigned)tdim1 & 0xFFFFu) << 16);
  unsigned w3 = ((unsigned)tdim1 >> 16) | ((unsigned)tile0 << 16);      // tile_dim0
  unsigned w4 = (unsigned)tile1;                      // tile_dim1 (tile_dim2=0)
  unsigned w5 = (unsigned)(stride0 & 0xFFFFFFFFull);  // dim0_stride[31:0]
  unsigned w6 = (unsigned)((stride0 >> 32) & 0xFFFFull);
  unsigned w7 = 0u;
  i32x8 g1 = {(int)w0, (int)w1, (int)w2, (int)w3,
              (int)w4, (int)w5, (int)w6, (int)w7};
  i32x4 z4 = (i32x4)0;
#if __clang_major__ >= 23
  i32x8 z8 = (i32x8)0;
  __builtin_amdgcn_tensor_load_to_lds(g0, g1, z4, z4, z8, 0);
#else
  __builtin_amdgcn_tensor_load_to_lds(g0, g1, z4, z4, 0);
#endif
}
#endif  // HAVE_TDM

__global__ void f32_to_bf16_kernel(const float* __restrict__ in,
                                   __bf16* __restrict__ out, int n) {
  int i = blockIdx.x * blockDim.x + threadIdx.x;
  int stride = gridDim.x * blockDim.x;
  for (; i < n; i += stride) out[i] = (__bf16)in[i];
}

// [K,N] f32 -> [N,K] bf16, per expert (blockIdx.z). Dims multiples of 32.
__global__ void transpose_cvt_kernel(const float* __restrict__ in,
                                     __bf16* __restrict__ out, int K, int N) {
  __shared__ float tile[32][33];
  const float* inE = in + (size_t)blockIdx.z * K * N;
  __bf16* outE = out + (size_t)blockIdx.z * K * N;
  int n0 = blockIdx.x * 32, k0 = blockIdx.y * 32;
  int tx = threadIdx.x, ty = threadIdx.y;  // 32 x 8
#pragma unroll
  for (int i = 0; i < 32; i += 8)
    tile[ty + i][tx] = inE[(size_t)(k0 + ty + i) * N + n0 + tx];
  __syncthreads();
#pragma unroll
  for (int i = 0; i < 32; i += 8)
    outE[(size_t)(n0 + ty + i) * K + k0 + tx] = (__bf16)tile[tx][ty + i];
}

__global__ void zero_f32_kernel(float* __restrict__ p, int n) {
  int i = blockIdx.x * blockDim.x + threadIdx.x;
  int stride = gridDim.x * blockDim.x;
  for (; i < n; i += stride) p[i] = 0.0f;
}

// EPI = 0 : out is bf16, epilogue = LeakyReLU(acc + bias)       (layers 1 & 2)
// EPI = 1 : out is f32,  epilogue = out += w[e,row]*(acc+bias)  (layer 3 + gate)
template <int EPI>
__global__ __launch_bounds__(256)
void moe_gemm_kernel(const __bf16* __restrict__ A,   // [M,K] bf16 row-major
                     const __bf16* __restrict__ Wt,  // [N,K] bf16 row-major
                     const float*  __restrict__ bias,// [N]
                     void* __restrict__ outp,        // [M,N]
                     int M, int N, int K,
                     const int*   __restrict__ topk_idx,   // [M,2] (EPI==1)
                     const float* __restrict__ topk_probs, // [M,2] (EPI==1)
                     int expert) {
  __shared__ __bf16 sA[2][BM * LP];  // 2 x 10 KB
  __shared__ __bf16 sW[2][BN * LP];  // 2 x 10 KB

  const int tid    = threadIdx.x;
  const int lane   = tid & 31;
  const int waveId = tid >> 5;       // 0..7
  const int waveM  = waveId >> 2;    // 0..1  -> 64 rows
  const int waveN  = waveId & 3;     // 0..3  -> 32 cols
  const int half   = lane >> 4;      // 0..1
  const int ln     = lane & 15;

  const int blockN0 = blockIdx.x * BN;
  const int blockM0 = blockIdx.y * BM;
  const int kSteps  = K / BK;

#if HAVE_TDM
  unsigned ldsA[2] = {lds_off(&sA[0][0]), lds_off(&sA[1][0])};
  unsigned ldsW[2] = {lds_off(&sW[0][0]), lds_off(&sW[1][0])};
#endif

  v8f acc[4][2];
#pragma unroll
  for (int i = 0; i < 4; ++i)
#pragma unroll
    for (int j = 0; j < 2; ++j)
#pragma unroll
      for (int r = 0; r < 8; ++r) acc[i][j][r] = 0.0f;

  auto stage = [&](int s, int buf) {
#if HAVE_TDM
    if (waveId == 0) {
      tdm_load_2d(A + (size_t)blockM0 * K + (size_t)s * BK, ldsA[buf],
                  K, M, (unsigned long long)K, BK, BM);
      tdm_load_2d(Wt + (size_t)blockN0 * K + (size_t)s * BK, ldsW[buf],
                  K, N, (unsigned long long)K, BK, BN);
    }
#else
    const __bf16* Ag = A + (size_t)blockM0 * K + (size_t)s * BK;
    const __bf16* Wg = Wt + (size_t)blockN0 * K + (size_t)s * BK;
#pragma unroll
    for (int j = 0; j < 2; ++j) {
      int chunk = tid * 2 + j;       // 0..511
      int row   = chunk >> 2;        // 0..127
      int seg   = (chunk & 3) << 3;  // 0,8,16,24 elements
      *(v8bf*)&sA[buf][row * LP + seg] = *(const v8bf*)(Ag + (size_t)row * K + seg);
      *(v8bf*)&sW[buf][row * LP + seg] = *(const v8bf*)(Wg + (size_t)row * K + seg);
    }
#endif
  };

  stage(0, 0);

  for (int s = 0; s < kSteps; ++s) {
    const int cur = s & 1;
#if HAVE_TDM
    __builtin_amdgcn_s_wait_tensorcnt(0);  // own TDM (tile s) done; nop for waves 1-7
#endif
    __syncthreads();                       // tile s visible to all waves
    if (s + 1 < kSteps) stage(s + 1, cur ^ 1);

    const __bf16* As = &sA[cur][0];
    const __bf16* Ws = &sW[cur][0];

    v16bf afrag[4];
#pragma unroll
    for (int mt = 0; mt < 4; ++mt) {
      const __bf16* base = As + (waveM * 64 + mt * 16 + ln) * LP;
      v8bf lo = *(const v8bf*)(base + half * 8);       // K = half*8 + 0..7
      v8bf hi = *(const v8bf*)(base + 16 + half * 8);  // K = 16 + half*8 + 0..7
      afrag[mt] = __builtin_shufflevector(lo, hi, 0, 1, 2, 3, 4, 5, 6, 7,
                                          8, 9, 10, 11, 12, 13, 14, 15);
    }
    v16bf bfrag[2];
#pragma unroll
    for (int nt = 0; nt < 2; ++nt) {
      const __bf16* base = Ws + (waveN * 32 + nt * 16 + ln) * LP;
      v8bf lo = *(const v8bf*)(base + half * 8);       // K = half*8 + 0..7
      v8bf hi = *(const v8bf*)(base + 16 + half * 8);  // K = 16 + half*8 + 0..7
      bfrag[nt] = __builtin_shufflevector(lo, hi, 0, 1, 2, 3, 4, 5, 6, 7,
                                          8, 9, 10, 11, 12, 13, 14, 15);
    }

#pragma unroll
    for (int mt = 0; mt < 4; ++mt)
#pragma unroll
      for (int nt = 0; nt < 2; ++nt)
        acc[mt][nt] = __builtin_amdgcn_wmma_f32_16x16x32_bf16(
            false, afrag[mt], false, bfrag[nt], (short)0, acc[mt][nt],
            false, false);
  }

  // --- epilogue --- (C/D layout: VGPR r, lane -> M = r + half*8, N = ln)
  if (EPI == 0) {
    __bf16* Hout = (__bf16*)outp;
#pragma unroll
    for (int mt = 0; mt < 4; ++mt) {
#pragma unroll
      for (int nt = 0; nt < 2; ++nt) {
        int col = blockN0 + waveN * 32 + nt * 16 + ln;
        float bv = bias[col];
#pragma unroll
        for (int r = 0; r < 8; ++r) {
          int row = blockM0 + waveM * 64 + mt * 16 + half * 8 + r;
          float v = acc[mt][nt][r] + bv;
          v = v > 0.0f ? v : 0.2f * v;  // LeakyReLU(0.2)
          Hout[(size_t)row * N + col] = (__bf16)v;
        }
      }
    }
  } else {
    float* out = (float*)outp;
#pragma unroll
    for (int mt = 0; mt < 4; ++mt) {
      float wrow[8];
#pragma unroll
      for (int r = 0; r < 8; ++r) {
        int row = blockM0 + waveM * 64 + mt * 16 + half * 8 + r;
        int   i0 = topk_idx[row * 2 + 0], i1 = topk_idx[row * 2 + 1];
        float p0 = topk_probs[row * 2 + 0], p1 = topk_probs[row * 2 + 1];
        wrow[r] = (i0 == expert ? p0 : 0.0f) + (i1 == expert ? p1 : 0.0f);
      }
#pragma unroll
      for (int nt = 0; nt < 2; ++nt) {
        int col = blockN0 + waveN * 32 + nt * 16 + ln;
        float bv = bias[col];
#pragma unroll
        for (int r = 0; r < 8; ++r) {
          int row = blockM0 + waveM * 64 + mt * 16 + half * 8 + r;
          out[(size_t)row * N + col] += wrow[r] * (acc[mt][nt][r] + bv);
        }
      }
    }
  }
}

extern "C" void kernel_launch(void* const* d_in, const int* in_sizes, int n_in,
                              void* d_out, int out_size, void* d_ws,
                              size_t ws_size, hipStream_t stream) {
  (void)in_sizes; (void)n_in; (void)out_size; (void)ws_size;

  enum { B_ = 16384, IN_ = 2048, D_ = 1024, E_ = 3, H1_ = 1024, H2_ = 512 };

  const float* x          = (const float*)d_in[0];
  // d_in[1] = gate (unused by reference)
  const float* topk_probs = (const float*)d_in[2];
  const int*   topk_idx   = (const int*)d_in[3];
  const float* W1 = (const float*)d_in[4];
  const float* b1 = (const float*)d_in[5];
  const float* W2 = (const float*)d_in[6];
  const float* b2 = (const float*)d_in[7];
  const float* W3 = (const float*)d_in[8];
  const float* b3 = (const float*)d_in[9];
  float* out = (float*)d_out;

  // workspace carve-up (~136 MB)
  char* ws = (char*)d_ws;
  __bf16* Xb   = (__bf16*)ws; ws += (size_t)B_ * IN_ * 2;        // 64 MB
  __bf16* W1T  = (__bf16*)ws; ws += (size_t)E_ * IN_ * H1_ * 2;  // 12 MB [E][H1][IN]
  __bf16* W2T  = (__bf16*)ws; ws += (size_t)E_ * H1_ * H2_ * 2;  //  3 MB [E][H2][H1]
  __bf16* W3T  = (__bf16*)ws; ws += (size_t)E_ * H2_ * D_ * 2;   //  3 MB [E][D][H2]
  __bf16* H1b  = (__bf16*)ws; ws += (size_t)B_ * H1_ * 2;        // 32 MB
  __bf16* H2b  = (__bf16*)ws; ws += (size_t)B_ * H2_ * 2;        // 16 MB

  zero_f32_kernel<<<2048, 256, 0, stream>>>(out, B_ * D_);
  f32_to_bf16_kernel<<<4096, 256, 0, stream>>>(x, Xb, B_ * IN_);

  dim3 tb(32, 8);
  transpose_cvt_kernel<<<dim3(H1_ / 32, IN_ / 32, E_), tb, 0, stream>>>(W1, W1T, IN_, H1_);
  transpose_cvt_kernel<<<dim3(H2_ / 32, H1_ / 32, E_), tb, 0, stream>>>(W2, W2T, H1_, H2_);
  transpose_cvt_kernel<<<dim3(D_ / 32, H2_ / 32, E_), tb, 0, stream>>>(W3, W3T, H2_, D_);

  for (int e = 0; e < E_; ++e) {
    moe_gemm_kernel<0><<<dim3(H1_ / BN, B_ / BM), 256, 0, stream>>>(
        Xb, W1T + (size_t)e * IN_ * H1_, b1 + e * H1_, H1b,
        B_, H1_, IN_, nullptr, nullptr, 0);
    moe_gemm_kernel<0><<<dim3(H2_ / BN, B_ / BM), 256, 0, stream>>>(
        H1b, W2T + (size_t)e * H1_ * H2_, b2 + e * H2_, H2b,
        B_, H2_, H1_, nullptr, nullptr, 0);
    moe_gemm_kernel<1><<<dim3(D_ / BN, B_ / BM), 256, 0, stream>>>(
        H2b, W3T + (size_t)e * H2_ * D_, b3 + e * D_, out,
        B_, D_, H2_, topk_idx, topk_probs, e);
  }
}
